// Residual_Comb_Conv_33071248179239
// MI455X (gfx1250) — matrix operand order repaired and checked by reading, not compiled, not run on hardware
//
#include <hip/hip_runtime.h>
#include <hip/hip_bf16.h>

// ---------------------------------------------------------------------------
// Residual comb-conv: out = Conv(BN2,relu,gather(Conv(BN1,relu,gather(x))))
//                         + Conv(BN3,relu,gather(x))
// B=2048, C_IN=C_MID=64, C_OUT=128, R=60, K=13, contraction length 832.
// One wave per batch element; all matrix work on v_wmma_f32_16x16x32_f16.
// ---------------------------------------------------------------------------

#define B_N     2048
#define R_N     60
#define K_N     13
#define CK_N    832            // 64*13
#define KC_N    26             // 832 / 32

typedef __attribute__((ext_vector_type(16))) _Float16       v16h;
typedef __attribute__((ext_vector_type(8)))  float          v8f;
typedef __attribute__((ext_vector_type(8)))  unsigned short u16x8;

// workspace layout (bytes)
#define WS_WT1   0          // 64*832  f16 = 106496
#define WS_WT2   106496     // 128*832 f16 = 212992
#define WS_WT3   319488     // 128*832 f16 = 212992
#define WS_IDX   532480     // 60*832  u16 =  99840
#define WS_BN    632320     // 6*64 f32 = 1536
// total ~634 KB

__device__ __forceinline__ unsigned int pack2(_Float16 a, _Float16 b) {
    unsigned short ua = __builtin_bit_cast(unsigned short, a);
    unsigned short ub = __builtin_bit_cast(unsigned short, b);
    return (unsigned int)ua | ((unsigned int)ub << 16);
}

__device__ __forceinline__ _Float16 bnrelu16(float x, float inv, float sh) {
    return (_Float16)fmaxf(fmaf(x, inv, sh), 0.0f);
}

// Build one 16x32 f16 A-fragment for this lane by gathering from LDS
// activations through the precomputed index table.
// idxrow points at idx[mrow*832 + kc*32 + (lane>=16 ? 8 : 0)].
__device__ __forceinline__ v16h build_afrag(const unsigned short* __restrict__ idxrow,
                                            const unsigned short* aP) {
    u16x8 i0 = *(const u16x8*)(idxrow);        // K elements 0..7 of this half
    u16x8 i1 = *(const u16x8*)(idxrow + 16);   // K elements 16..23 of this half
    v16h a;
#pragma unroll
    for (int i = 0; i < 8; ++i) {
        a[i]     = __builtin_bit_cast(_Float16, aP[i0[i]]);
        a[i + 8] = __builtin_bit_cast(_Float16, aP[i1[i]]);
    }
    return a;
}

// ---------------------------------------------------------------------------
// Prep kernel: f32->f16 weights (layout [o][ck], ck = c*13+k == source flat
// order), gather index table, folded BN affines (c1 folded into BN2 shift).
// ---------------------------------------------------------------------------
__global__ void prep_kernel(const float* __restrict__ W1, const float* __restrict__ W2,
                            const float* __restrict__ W3, const int* __restrict__ nei,
                            const float* __restrict__ g1, const float* __restrict__ b1,
                            const float* __restrict__ m1, const float* __restrict__ v1,
                            const float* __restrict__ g2, const float* __restrict__ b2,
                            const float* __restrict__ m2, const float* __restrict__ v2,
                            const float* __restrict__ c1,
                            const float* __restrict__ g3, const float* __restrict__ b3,
                            const float* __restrict__ m3, const float* __restrict__ v3,
                            _Float16* __restrict__ wt1, _Float16* __restrict__ wt2,
                            _Float16* __restrict__ wt3, unsigned short* __restrict__ idxt,
                            float* __restrict__ bn) {
    int tid    = blockIdx.x * blockDim.x + threadIdx.x;
    int stride = gridDim.x * blockDim.x;

    for (int e = tid; e < 64 * CK_N; e += stride)  wt1[e] = (_Float16)W1[e];
    for (int e = tid; e < 128 * CK_N; e += stride) wt2[e] = (_Float16)W2[e];
    for (int e = tid; e < 128 * CK_N; e += stride) wt3[e] = (_Float16)W3[e];

    for (int e = tid; e < R_N * CK_N; e += stride) {
        int r  = e / CK_N;
        int ck = e - r * CK_N;
        int c  = ck / K_N;
        int k  = ck - c * K_N;
        idxt[e] = (unsigned short)(c * R_N + nei[r * K_N + k]);
    }

    if (tid < 64) {
        float i1 = g1[tid] * rsqrtf(v1[tid] + 1e-5f);
        bn[tid]       = i1;
        bn[64 + tid]  = b1[tid] - m1[tid] * i1;
        float i2 = g2[tid] * rsqrtf(v2[tid] + 1e-5f);
        bn[128 + tid] = i2;
        bn[192 + tid] = b2[tid] + (c1[tid] - m2[tid]) * i2;   // conv bias c1 folded in
        float i3 = g3[tid] * rsqrtf(v3[tid] + 1e-5f);
        bn[256 + tid] = i3;
        bn[320 + tid] = b3[tid] - m3[tid] * i3;
    }
}

// ---------------------------------------------------------------------------
// Fused main kernel: 4 waves / block, one batch element per wave.
// LDS per wave: a1/a2 buffer (64ch x 60 f16 = 7.5KB) + a3 buffer (7.5KB).
// ---------------------------------------------------------------------------
__global__ __launch_bounds__(128)
void comb_conv_kernel(const float* __restrict__ x,
                      const unsigned short* __restrict__ idxt,
                      const _Float16* __restrict__ wt1,
                      const _Float16* __restrict__ wt2,
                      const _Float16* __restrict__ wt3,
                      const float* __restrict__ bn,
                      const float* __restrict__ c2,
                      const float* __restrict__ c3,
                      float* __restrict__ out) {
    extern __shared__ unsigned short smem[];         // 4 waves * 7680 u16
    const int lane = threadIdx.x & 31;
    const int wave = threadIdx.x >> 5;
    const int b    = blockIdx.x * 4 + wave;

    unsigned short* aA = smem + wave * 7680;         // a1, later a2  (c*60 + r)
    unsigned short* aB = aA + 3840;                  // a3

    const float* inv1 = bn;        const float* sh1 = bn + 64;
    const float* inv2 = bn + 128;  const float* sh2 = bn + 192;
    const float* inv3 = bn + 256;  const float* sh3 = bn + 320;

    const int half = lane >> 4;    // which 16-lane half of the wave
    const int l15  = lane & 15;

    // ---- Phase 0: a1 = relu(BN1(x)), a3 = relu(BN3(x)) as f16 in LDS -------
    const float* xb = x + (size_t)b * (64 * R_N);
#pragma unroll 1
    for (int t = 0; t < 30; ++t) {                   // 960 float4 / 32 lanes
        int e4 = t * 32 + lane;
        int c  = e4 / 15;                            // 15 float4 per 60-row
        int r  = (e4 - c * 15) * 4;
        float4 xv = ((const float4*)xb)[e4];
        float i1v = inv1[c], s1v = sh1[c];
        float i3v = inv3[c], s3v = sh3[c];
        uint2 p1, p3;
        p1.x = pack2(bnrelu16(xv.x, i1v, s1v), bnrelu16(xv.y, i1v, s1v));
        p1.y = pack2(bnrelu16(xv.z, i1v, s1v), bnrelu16(xv.w, i1v, s1v));
        p3.x = pack2(bnrelu16(xv.x, i3v, s3v), bnrelu16(xv.y, i3v, s3v));
        p3.y = pack2(bnrelu16(xv.z, i3v, s3v), bnrelu16(xv.w, i3v, s3v));
        *(uint2*)(aA + c * R_N + r) = p1;
        *(uint2*)(aB + c * R_N + r) = p3;
    }
    // LDS ops from the same wave are in-order; no barrier needed (private slice).

    int mrow[4];
#pragma unroll
    for (int mt = 0; mt < 4; ++mt) {
        int m = mt * 16 + l15;
        mrow[mt] = (m > 59) ? 59 : m;                // clamp padded rows
    }

    // ---- Stage 1: f1[64 x 60] = gather(a1) x W1  -> BN2+ReLU -> a2 ---------
    v8f acc1[4][4];
#pragma unroll
    for (int mt = 0; mt < 4; ++mt)
#pragma unroll
        for (int nt = 0; nt < 4; ++nt) acc1[mt][nt] = (v8f)0.0f;

#pragma unroll 1
    for (int kc = 0; kc < KC_N; ++kc) {
        const int kbase = kc * 32;
        v16h af[4];
#pragma unroll
        for (int mt = 0; mt < 4; ++mt)
            af[mt] = build_afrag(idxt + mrow[mt] * CK_N + kbase + half * 8, aA);
#pragma unroll
        for (int nt = 0; nt < 4; ++nt) {
            const _Float16* wp = wt1 + (nt * 16 + l15) * CK_N + kbase + half * 8;
            v16h bf = __builtin_bit_cast(v16h, __builtin_shufflevector(*(const u16x8*)wp, *(const u16x8*)(wp + 16), 0, 1, 2, 3, 4, 5, 6, 7, 8, 9, 10, 11, 12, 13, 14, 15));
#pragma unroll
            for (int mt = 0; mt < 4; ++mt)
                acc1[mt][nt] = __builtin_amdgcn_wmma_f32_16x16x32_f16(
                    false, af[mt], false, bf, (short)0, acc1[mt][nt], false, false);
        }
    }

    // write a2 = relu(BN2(f1 + c1)) back into aA (a1 is dead)
#pragma unroll
    for (int nt = 0; nt < 4; ++nt) {
        int o = nt * 16 + l15;
        float iv = inv2[o], sv = sh2[o];
#pragma unroll
        for (int mt = 0; mt < 4; ++mt) {
#pragma unroll
            for (int j = 0; j < 4; ++j) {
                int m0 = mt * 16 + half * 8 + 2 * j;   // D: VGPR j2 -> m = base+j2
                if (m0 < 60) {
                    unsigned int p = pack2(bnrelu16(acc1[mt][nt][2 * j],     iv, sv),
                                           bnrelu16(acc1[mt][nt][2 * j + 1], iv, sv));
                    *(unsigned int*)(aA + o * R_N + m0) = p;
                }
            }
        }
    }

    // ---- Stage 2 + 3: out = gather(a2) x W2 + gather(a3) x W3 + c2 + c3 ----
    v8f acc2[4][8];
#pragma unroll
    for (int mt = 0; mt < 4; ++mt)
#pragma unroll
        for (int nt = 0; nt < 8; ++nt) acc2[mt][nt] = (v8f)0.0f;

#pragma unroll 1
    for (int kc = 0; kc < KC_N; ++kc) {
        const int kbase = kc * 32;
        v16h af[4];
#pragma unroll
        for (int mt = 0; mt < 4; ++mt)
            af[mt] = build_afrag(idxt + mrow[mt] * CK_N + kbase + half * 8, aA);
#pragma unroll
        for (int nt = 0; nt < 8; ++nt) {
            const _Float16* wp = wt2 + (nt * 16 + l15) * CK_N + kbase + half * 8;
            v16h bf = __builtin_bit_cast(v16h, __builtin_shufflevector(*(const u16x8*)wp, *(const u16x8*)(wp + 16), 0, 1, 2, 3, 4, 5, 6, 7, 8, 9, 10, 11, 12, 13, 14, 15));
#pragma unroll
            for (int mt = 0; mt < 4; ++mt)
                acc2[mt][nt] = __builtin_amdgcn_wmma_f32_16x16x32_f16(
                    false, af[mt], false, bf, (short)0, acc2[mt][nt], false, false);
        }
#pragma unroll
        for (int mt = 0; mt < 4; ++mt)
            af[mt] = build_afrag(idxt + mrow[mt] * CK_N + kbase + half * 8, aB);
#pragma unroll
        for (int nt = 0; nt < 8; ++nt) {
            const _Float16* wp = wt3 + (nt * 16 + l15) * CK_N + kbase + half * 8;
            v16h bf = __builtin_bit_cast(v16h, __builtin_shufflevector(*(const u16x8*)wp, *(const u16x8*)(wp + 16), 0, 1, 2, 3, 4, 5, 6, 7, 8, 9, 10, 11, 12, 13, 14, 15));
#pragma unroll
            for (int mt = 0; mt < 4; ++mt)
                acc2[mt][nt] = __builtin_amdgcn_wmma_f32_16x16x32_f16(
                    false, af[mt], false, bf, (short)0, acc2[mt][nt], false, false);
        }
    }

    // ---- Epilogue: add conv biases, store f32 [b][o][r] --------------------
    float* ob = out + (size_t)b * (128 * R_N);
#pragma unroll
    for (int nt = 0; nt < 8; ++nt) {
        int o = nt * 16 + l15;
        float cb = c2[o] + c3[o];
#pragma unroll
        for (int mt = 0; mt < 4; ++mt) {
            int m0 = mt * 16 + half * 8;              // rows m0..m0+7 in this lane
            float4 lo, hi;
            lo.x = acc2[mt][nt][0] + cb;  lo.y = acc2[mt][nt][1] + cb;
            lo.z = acc2[mt][nt][2] + cb;  lo.w = acc2[mt][nt][3] + cb;
            *(float4*)(ob + o * R_N + m0) = lo;       // m0 <= 56, always valid
            if (m0 + 4 < 60) {
                hi.x = acc2[mt][nt][4] + cb;  hi.y = acc2[mt][nt][5] + cb;
                hi.z = acc2[mt][nt][6] + cb;  hi.w = acc2[mt][nt][7] + cb;
                *(float4*)(ob + o * R_N + m0 + 4) = hi;
            }
        }
    }
}

// ---------------------------------------------------------------------------
extern "C" void kernel_launch(void* const* d_in, const int* in_sizes, int n_in,
                              void* d_out, int out_size, void* d_ws, size_t ws_size,
                              hipStream_t stream) {
    const float* x   = (const float*)d_in[0];
    const int*   nei = (const int*)d_in[1];
    const float* g1 = (const float*)d_in[2],  *b1 = (const float*)d_in[3];
    const float* m1 = (const float*)d_in[4],  *v1 = (const float*)d_in[5];
    const float* W1 = (const float*)d_in[6],  *c1 = (const float*)d_in[7];
    const float* g2 = (const float*)d_in[8],  *b2 = (const float*)d_in[9];
    const float* m2 = (const float*)d_in[10], *v2 = (const float*)d_in[11];
    const float* W2 = (const float*)d_in[12], *c2 = (const float*)d_in[13];
    const float* g3 = (const float*)d_in[14], *b3 = (const float*)d_in[15];
    const float* m3 = (const float*)d_in[16], *v3 = (const float*)d_in[17];
    const float* W3 = (const float*)d_in[18], *c3 = (const float*)d_in[19];

    char* ws = (char*)d_ws;
    _Float16*       wt1  = (_Float16*)(ws + WS_WT1);
    _Float16*       wt2  = (_Float16*)(ws + WS_WT2);
    _Float16*       wt3  = (_Float16*)(ws + WS_WT3);
    unsigned short* idxt = (unsigned short*)(ws + WS_IDX);
    float*          bn   = (float*)(ws + WS_BN);

    prep_kernel<<<208, 256, 0, stream>>>(W1, W2, W3, nei,
                                         g1, b1, m1, v1,
                                         g2, b2, m2, v2, c1,
                                         g3, b3, m3, v3,
                                         wt1, wt2, wt3, idxt, bn);

    // 512 blocks * 4 waves = 2048 batch elements; 61440 B dynamic LDS / block
    comb_conv_kernel<<<B_N / 4, 128, 4 * 7680 * sizeof(unsigned short), stream>>>(
        x, idxt, wt1, wt2, wt3, bn, c2, c3, (float*)d_out);
}
